// SchNetWrapper_27041114095741
// MI455X (gfx1250) — matrix operand order repaired and checked by reading, not compiled, not run on hardware
//
#include <hip/hip_runtime.h>
#include <hip/hip_bf16.h>
#include <math.h>

#define HIDDEN   128
#define NGAUSS   50
#define KG       64        // NUM_GAUSS padded to WMMA K granule
#define NGRAPHS  100
#define LLAYERS  6

typedef __attribute__((ext_vector_type(16))) _Float16 v16h;
typedef __attribute__((ext_vector_type(8)))  _Float16 v8h;
typedef __attribute__((ext_vector_type(8)))  float    v8f;

enum { F_BIAS = 1, F_SSP = 2, F_RESID = 4, F_AUXDSSP = 8, F_ROWSCALE = 16 };

__device__ __forceinline__ float sspf(float x) {
  // shifted softplus: log(1+e^x) - log(2), numerically stable
  return fmaxf(x, 0.f) + log1pf(__expf(-fabsf(x))) - 0.69314718055994531f;
}

// ---- typed helpers (overload-resolved at compile time) ----------------------
__device__ __forceinline__ float ldf(const float* p, size_t i)    { return p[i]; }
__device__ __forceinline__ float ldf(const _Float16* p, size_t i) { return (float)p[i]; }
__device__ __forceinline__ void  stf(float* p, size_t i, float v)    { p[i] = v; }
__device__ __forceinline__ void  stf(_Float16* p, size_t i, float v) { p[i] = (_Float16)v; }

// A fragment loaders: 16x32 f16 WMMA A layout, row-major source, lda % 8 == 0.
__device__ __forceinline__ v16h load_a(const float* arow, int k0, bool rv) {
  float4 f0 = make_float4(0.f,0.f,0.f,0.f), f1 = f0, f2 = f0, f3 = f0;
  if (rv) {
    f0 = *reinterpret_cast<const float4*>(arow + k0);
    f1 = *reinterpret_cast<const float4*>(arow + k0 + 4);
    f2 = *reinterpret_cast<const float4*>(arow + k0 + 16);
    f3 = *reinterpret_cast<const float4*>(arow + k0 + 20);
  }
  v16h a;
  a[0]=(_Float16)f0.x;  a[1]=(_Float16)f0.y;  a[2]=(_Float16)f0.z;  a[3]=(_Float16)f0.w;
  a[4]=(_Float16)f1.x;  a[5]=(_Float16)f1.y;  a[6]=(_Float16)f1.z;  a[7]=(_Float16)f1.w;
  a[8]=(_Float16)f2.x;  a[9]=(_Float16)f2.y;  a[10]=(_Float16)f2.z; a[11]=(_Float16)f2.w;
  a[12]=(_Float16)f3.x; a[13]=(_Float16)f3.y; a[14]=(_Float16)f3.z; a[15]=(_Float16)f3.w;
  return a;
}
__device__ __forceinline__ v16h load_a_rs(const float* arow, int k0, bool rv, float rs) {
  float4 f0 = make_float4(0.f,0.f,0.f,0.f), f1 = f0, f2 = f0, f3 = f0;
  if (rv) {
    f0 = *reinterpret_cast<const float4*>(arow + k0);
    f1 = *reinterpret_cast<const float4*>(arow + k0 + 4);
    f2 = *reinterpret_cast<const float4*>(arow + k0 + 16);
    f3 = *reinterpret_cast<const float4*>(arow + k0 + 20);
  }
  v16h a;
  a[0]=(_Float16)(f0.x*rs);  a[1]=(_Float16)(f0.y*rs);
  a[2]=(_Float16)(f0.z*rs);  a[3]=(_Float16)(f0.w*rs);
  a[4]=(_Float16)(f1.x*rs);  a[5]=(_Float16)(f1.y*rs);
  a[6]=(_Float16)(f1.z*rs);  a[7]=(_Float16)(f1.w*rs);
  a[8]=(_Float16)(f2.x*rs);  a[9]=(_Float16)(f2.y*rs);
  a[10]=(_Float16)(f2.z*rs); a[11]=(_Float16)(f2.w*rs);
  a[12]=(_Float16)(f3.x*rs); a[13]=(_Float16)(f3.y*rs);
  a[14]=(_Float16)(f3.z*rs); a[15]=(_Float16)(f3.w*rs);
  return a;
}
__device__ __forceinline__ v16h load_a(const _Float16* arow, int k0, bool rv) {
  v8h z = {(_Float16)0,(_Float16)0,(_Float16)0,(_Float16)0,
           (_Float16)0,(_Float16)0,(_Float16)0,(_Float16)0};
  v8h h0 = z, h1 = z;
  if (rv) {
    h0 = *reinterpret_cast<const v8h*>(arow + k0);
    h1 = *reinterpret_cast<const v8h*>(arow + k0 + 16);
  }
  v16h a;
#pragma unroll
  for (int i = 0; i < 8; ++i) { a[i] = h0[i]; a[8+i] = h1[i]; }
  return a;
}
__device__ __forceinline__ v16h load_a_rs(const _Float16* arow, int k0, bool rv, float rs) {
  v8h z = {(_Float16)0,(_Float16)0,(_Float16)0,(_Float16)0,
           (_Float16)0,(_Float16)0,(_Float16)0,(_Float16)0};
  v8h h0 = z, h1 = z;
  if (rv) {
    h0 = *reinterpret_cast<const v8h*>(arow + k0);
    h1 = *reinterpret_cast<const v8h*>(arow + k0 + 16);
  }
  v16h a;
#pragma unroll
  for (int i = 0; i < 8; ++i) {
    a[i]   = (_Float16)((float)h0[i] * rs);
    a[8+i] = (_Float16)((float)h1[i] * rs);
  }
  return a;
}

// ---------------------------------------------------------------------------
// Generic WMMA GEMM: C[M, NT*16] = epilogue(A[M,lda] @ Bfrag + ...)
// Bfrag: prepacked f16 fragments, layout [(kt*NT+nt)*32 + lane][16 halves]
// ---------------------------------------------------------------------------
template<int KT, int NT, int FLAGS, typename TA, typename TC, typename TX>
__global__ __launch_bounds__(128)
void gemm_wmma(const TA* __restrict__ A, int lda,
               const _Float16* __restrict__ Bf,
               const float* __restrict__ bias,
               const float* __restrict__ resid,
               const TX* __restrict__ aux,
               const float* __restrict__ rscale,
               TC* __restrict__ C, int M)
{
  const int ldc  = NT * 16;
  const int lane = threadIdx.x & 31;
  const int wave = threadIdx.x >> 5;
  const int m0   = (blockIdx.x * 4 + wave) << 4;
  if (m0 >= M) return;

  const int  arow_i = m0 + (lane & 15);
  const bool rv     = arow_i < M;
  const TA*  arow   = A + (size_t)arow_i * lda;
  float rs = 1.f;
  if (FLAGS & F_ROWSCALE) rs = rv ? rscale[arow_i] : 0.f;
  const int koff = (lane >> 4) << 3;   // lanes 16..31 hold K+8 block

  // speculative prefetch of the next 64-row block of A
  __builtin_prefetch((const void*)(A + (size_t)(arow_i + 64) * lda), 0, 1);

  v8f zerov = {0.f,0.f,0.f,0.f,0.f,0.f,0.f,0.f};
  v8f acc[NT];
#pragma unroll
  for (int nt = 0; nt < NT; ++nt) acc[nt] = zerov;

#pragma unroll
  for (int kt = 0; kt < KT; ++kt) {
    const int k0 = kt * 32 + koff;
    v16h a;
    if (FLAGS & F_ROWSCALE) a = load_a_rs(arow, k0, rv, rs);
    else                    a = load_a(arow, k0, rv);
#pragma unroll
    for (int nt = 0; nt < NT; ++nt) {
      v16h b = *reinterpret_cast<const v16h*>(Bf + (size_t)(((kt*NT + nt) << 5) + lane) * 16);
      acc[nt] = __builtin_amdgcn_wmma_f32_16x16x32_f16(false, a, false, b,
                                                       (short)0, acc[nt], false, false);
    }
  }

  const int rbase = m0 + ((lane >> 4) << 3);
#pragma unroll
  for (int nt = 0; nt < NT; ++nt) {
    const int col = (nt << 4) + (lane & 15);
    float bv = 0.f;
    if (FLAGS & F_BIAS) bv = bias[col];
#pragma unroll
    for (int r = 0; r < 8; ++r) {
      const int row = rbase + r;
      if (row < M) {
        float v = acc[nt][r] + bv;
        if (FLAGS & F_SSP)     v = sspf(v);
        if (FLAGS & F_AUXDSSP) v *= (1.f - 0.5f * __expf(-ldf(aux, (size_t)row*ldc + col)));
        if (FLAGS & F_RESID)   v += resid[(size_t)row*ldc + col];
        stf(C, (size_t)row*ldc + col, v);
      }
    }
  }
}

// ---------------------------------------------------------------------------
// Prepack f32 weight matrix into WMMA B fragments (optionally transposed).
// Logical B is [Kp x Np]; B[k][n] = transpose ? src[n][k] : src[k][n], 0-padded.
// ---------------------------------------------------------------------------
__global__ void prepack_kernel(const float* __restrict__ src, int srcRows, int srcCols,
                               int Kp, int Np, int transpose, _Float16* __restrict__ dst)
{
  int idx = blockIdx.x * blockDim.x + threadIdx.x;
  int total = Kp * Np;
  if (idx >= total) return;
  int frag = idx >> 9;        // 32 lanes * 16 halves
  int rem  = idx & 511;
  int lane = rem >> 4;
  int e    = rem & 15;
  int ntiles = Np >> 4;
  int kt = frag / ntiles;
  int nt = frag - kt * ntiles;
  int k = kt * 32 + ((lane >> 4) << 3) + (e & 7) + ((e >> 3) << 4);
  int n = nt * 16 + (lane & 15);
  float v = 0.f;
  if (!transpose) { if (k < srcRows && n < srcCols) v = src[(size_t)k * srcCols + n]; }
  else            { if (k < srcCols && n < srcRows) v = src[(size_t)n * srcCols + k]; }
  dst[idx] = (_Float16)v;
}

__global__ void zero_kernel(float* __restrict__ p, size_t n) {
  size_t i = (size_t)blockIdx.x * blockDim.x + threadIdx.x;
  if (i < n) p[i] = 0.f;
}

__global__ void embed_kernel(const float* __restrict__ emb, const int* __restrict__ z,
                             int N, float* __restrict__ h) {
  long long i = (long long)blockIdx.x * blockDim.x + threadIdx.x;
  if (i >= (long long)N * HIDDEN) return;
  int n = (int)(i >> 7), f = (int)(i & 127);
  h[i] = emb[(size_t)z[n] * HIDDEN + f];
}

__global__ void edge_geom_kernel(const float* __restrict__ pos,
                                 const int* __restrict__ src, const int* __restrict__ dst,
                                 int E, _Float16* __restrict__ rbf,
                                 float* __restrict__ dist, float* __restrict__ ccv)
{
  int e = blockIdx.x * blockDim.x + threadIdx.x;
  if (e >= E) return;
  int s = src[e], d = dst[e];
  float dx = pos[3*s+0]-pos[3*d+0];
  float dy = pos[3*s+1]-pos[3*d+1];
  float dz = pos[3*s+2]-pos[3*d+2];
  float r = sqrtf(dx*dx + dy*dy + dz*dz);
  dist[e] = r;
  const float pic = 3.14159265358979323846f / 10.f;
  ccv[e] = 0.5f * (__cosf(r * pic) + 1.f);
  const float delta = 10.f / 49.f;
  const float coeff = -0.5f / (delta * delta);
#pragma unroll 8
  for (int g = 0; g < KG; ++g) {
    float v = 0.f;
    if (g < NGAUSS) { float t = r - (float)g * delta; v = __expf(coeff * t * t); }
    rbf[(size_t)e * KG + g] = (_Float16)v;
  }
}

__global__ void scatter_msg_kernel(const float* __restrict__ xf, const _Float16* __restrict__ t2,
                                   const float* __restrict__ ccv,
                                   const int* __restrict__ src, const int* __restrict__ dst,
                                   int E, float* __restrict__ agg)
{
  long long idx = (long long)blockIdx.x * blockDim.x + threadIdx.x;
  if (idx >= (long long)E * HIDDEN) return;
  int e = (int)(idx >> 7), f = (int)(idx & 127);
  float m = xf[(size_t)src[e] * HIDDEN + f] * (float)t2[idx] * ccv[e];
  atomicAdd(&agg[(size_t)dst[e] * HIDDEN + f], m);
}

__global__ void scatter_bwd_kernel(const float* __restrict__ gagg, const _Float16* __restrict__ t2,
                                   const float* __restrict__ ccv, const float* __restrict__ xf,
                                   const int* __restrict__ src, const int* __restrict__ dst,
                                   int E, float* __restrict__ gxf, _Float16* __restrict__ gW)
{
  long long idx = (long long)blockIdx.x * blockDim.x + threadIdx.x;
  if (idx >= (long long)E * HIDDEN) return;
  int e = (int)(idx >> 7), f = (int)(idx & 127);
  int sn = src[e], dn = dst[e];
  float ga = gagg[(size_t)dn * HIDDEN + f];
  atomicAdd(&gxf[(size_t)sn * HIDDEN + f], ga * (float)t2[idx] * ccv[e]);
  gW[idx] = (_Float16)(ga * xf[(size_t)sn * HIDDEN + f]);
}

__global__ __launch_bounds__(128)
void gcc_kernel(const _Float16* __restrict__ gW, const _Float16* __restrict__ t2,
                int E, float* __restrict__ gcc)
{
  int lane = threadIdx.x & 31;
  int e = blockIdx.x * 4 + (threadIdx.x >> 5);
  if (e >= E) return;
  const _Float16* a = gW + (size_t)e * HIDDEN;
  const _Float16* b = t2 + (size_t)e * HIDDEN;
  float s = 0.f;
#pragma unroll
  for (int j = 0; j < 4; ++j) s += (float)a[lane + 32*j] * (float)b[lane + 32*j];
#pragma unroll
  for (int off = 16; off; off >>= 1) s += __shfl_xor(s, off, 32);
  if (lane == 0) gcc[e] = s;
}

__global__ void edge_force_kernel(const _Float16* __restrict__ grbf, const _Float16* __restrict__ rbf,
                                  const float* __restrict__ dist, const float* __restrict__ gcc,
                                  const float* __restrict__ pos,
                                  const int* __restrict__ src, const int* __restrict__ dst,
                                  int E, float* __restrict__ force)
{
  int e = blockIdx.x * blockDim.x + threadIdx.x;
  if (e >= E) return;
  float r = dist[e];
  const float pic = 3.14159265358979323846f / 10.f;
  float gd = gcc[e] * (-0.5f * pic * __sinf(r * pic));
  const float delta = 10.f / 49.f;
  const float coeff = -0.5f / (delta * delta);
#pragma unroll 10
  for (int g = 0; g < NGAUSS; ++g) {
    gd += (float)grbf[(size_t)e*KG + g] * (float)rbf[(size_t)e*KG + g]
          * (2.f * coeff * (r - (float)g * delta));
  }
  int s = src[e], d = dst[e];
  float inv = (r > 1e-10f) ? (gd / r) : 0.f;
  float ux = (pos[3*s+0]-pos[3*d+0]) * inv;
  float uy = (pos[3*s+1]-pos[3*d+1]) * inv;
  float uz = (pos[3*s+2]-pos[3*d+2]) * inv;
  // forces = -grad(pos)
  atomicAdd(&force[3*s+0], -ux); atomicAdd(&force[3*s+1], -uy); atomicAdd(&force[3*s+2], -uz);
  atomicAdd(&force[3*d+0],  ux); atomicAdd(&force[3*d+1],  uy); atomicAdd(&force[3*d+2],  uz);
}

__global__ void energy_kernel(const float* __restrict__ u, const float* __restrict__ h2w,
                              const float* __restrict__ h2b, const int* __restrict__ batch,
                              int N, float* __restrict__ energy)
{
  int n = blockIdx.x * blockDim.x + threadIdx.x;
  if (n >= N) return;
  float s = h2b[0];
#pragma unroll 8
  for (int j = 0; j < 64; ++j) s += u[(size_t)n*64 + j] * h2w[j];
  atomicAdd(&energy[batch[n]], s);
}

__global__ void ghead_kernel(const float* __restrict__ u, const float* __restrict__ h2w,
                             int N, float* __restrict__ ghp)
{
  long long i = (long long)blockIdx.x * blockDim.x + threadIdx.x;
  if (i >= (long long)N * 64) return;
  int j = (int)(i & 63);
  ghp[i] = h2w[j] * (1.f - 0.5f * __expf(-u[i]));
}

// ---------------------------------------------------------------------------
extern "C" void kernel_launch(void* const* d_in, const int* in_sizes, int n_in,
                              void* d_out, int out_size, void* d_ws, size_t ws_size,
                              hipStream_t stream)
{
  (void)n_in; (void)ws_size;
  const float* pos  = (const float*)d_in[0];
  const int*   z    = (const int*)d_in[1];
  const int*   ei   = (const int*)d_in[2];
  const int*   batch= (const int*)d_in[3];
  const float* emb  = (const float*)d_in[4];
  const float* w1   = (const float*)d_in[5];
  const float* b1   = (const float*)d_in[6];
  const float* w2   = (const float*)d_in[7];
  const float* b2   = (const float*)d_in[8];
  const float* l1w  = (const float*)d_in[9];
  const float* l2w  = (const float*)d_in[10];
  const float* l2b  = (const float*)d_in[11];
  const float* pw   = (const float*)d_in[12];
  const float* pb   = (const float*)d_in[13];
  const float* h1w  = (const float*)d_in[14];
  const float* h1b  = (const float*)d_in[15];
  const float* h2w  = (const float*)d_in[16];
  const float* h2b  = (const float*)d_in[17];

  const int N = in_sizes[0] / 3;
  const int E = in_sizes[2] / 2;
  const int* esrc = ei;
  const int* edst = ei + E;

  float* out        = (float*)d_out;
  float* out_energy = out;               // [NGRAPHS]
  float* out_force  = out + NGRAPHS;     // [N,3]; stress+features remain zero

  // ---- workspace bump allocator ----
  char* wsp = (char*)d_ws;
  auto alloc = [&](size_t bytes) -> void* {
    void* p = (void*)wsp;
    wsp += (bytes + 255) & ~(size_t)255;
    return p;
  };

  // ---- weight fragment prepack ----
  _Float16* fragPool = (_Float16*)alloc(sizeof(_Float16) * (size_t)2 * 1024 * 1024);
  size_t fo = 0;
  auto pack = [&](const float* src, int r, int c, int Kp, int Np, int tr) -> _Float16* {
    _Float16* d = fragPool + fo;
    fo += (size_t)Kp * Np;
    int tot = Kp * Np;
    prepack_kernel<<<dim3((unsigned)((tot + 255) / 256)), dim3(256), 0, stream>>>(
        src, r, c, Kp, Np, tr, d);
    return d;
  };

  _Float16 *w1f[LLAYERS], *w2f[LLAYERS], *l1f[LLAYERS], *l2f[LLAYERS], *pf[LLAYERS];
  _Float16 *w1t[LLAYERS], *w2t[LLAYERS], *l1t[LLAYERS], *l2t[LLAYERS], *pt[LLAYERS];
  for (int l = 0; l < LLAYERS; ++l) {
    const float* w1l = w1  + (size_t)l * NGAUSS * HIDDEN;
    const float* w2l = w2  + (size_t)l * HIDDEN * HIDDEN;
    const float* l1l = l1w + (size_t)l * HIDDEN * HIDDEN;
    const float* l2l = l2w + (size_t)l * HIDDEN * HIDDEN;
    const float* pwl = pw  + (size_t)l * HIDDEN * HIDDEN;
    w1f[l] = pack(w1l, NGAUSS, HIDDEN, KG,     HIDDEN, 0);
    w2f[l] = pack(w2l, HIDDEN, HIDDEN, HIDDEN, HIDDEN, 0);
    l1f[l] = pack(l1l, HIDDEN, HIDDEN, HIDDEN, HIDDEN, 0);
    l2f[l] = pack(l2l, HIDDEN, HIDDEN, HIDDEN, HIDDEN, 0);
    pf[l]  = pack(pwl, HIDDEN, HIDDEN, HIDDEN, HIDDEN, 0);
    w1t[l] = pack(w1l, NGAUSS, HIDDEN, HIDDEN, KG,     1);
    w2t[l] = pack(w2l, HIDDEN, HIDDEN, HIDDEN, HIDDEN, 1);
    l1t[l] = pack(l1l, HIDDEN, HIDDEN, HIDDEN, HIDDEN, 1);
    l2t[l] = pack(l2l, HIDDEN, HIDDEN, HIDDEN, HIDDEN, 1);
    pt[l]  = pack(pwl, HIDDEN, HIDDEN, HIDDEN, HIDDEN, 1);
  }
  _Float16* h1f = pack(h1w, HIDDEN, 64, HIDDEN, 64, 0);
  _Float16* h1t = pack(h1w, HIDDEN, 64, 64, HIDDEN, 1);

  // ---- persistent buffers ----
  _Float16* rbfh  = (_Float16*)alloc(sizeof(_Float16) * (size_t)E * KG);
  _Float16* grbfh = (_Float16*)alloc(sizeof(_Float16) * (size_t)E * KG);
  _Float16* et1h  = (_Float16*)alloc(sizeof(_Float16) * (size_t)E * HIDDEN);
  _Float16* et2h  = (_Float16*)alloc(sizeof(_Float16) * (size_t)E * HIDDEN);
  _Float16* gWh   = (_Float16*)alloc(sizeof(_Float16) * (size_t)E * HIDDEN);
  float* dist = (float*)alloc(sizeof(float) * (size_t)E);
  float* ccv  = (float*)alloc(sizeof(float) * (size_t)E);
  float* gcc  = (float*)alloc(sizeof(float) * (size_t)E);
  float* hbuf = (float*)alloc(sizeof(float) * (size_t)(LLAYERS + 1) * N * HIDDEN);
  float* xfb  = (float*)alloc(sizeof(float) * (size_t)LLAYERS * N * HIDDEN);
  float* vb   = (float*)alloc(sizeof(float) * (size_t)LLAYERS * N * HIDDEN);
  float* ub   = (float*)alloc(sizeof(float) * (size_t)N * 64);
  float* ghp  = (float*)alloc(sizeof(float) * (size_t)N * 64);
  float* agg  = (float*)alloc(sizeof(float) * (size_t)N * HIDDEN);
  float* gh   = (float*)alloc(sizeof(float) * (size_t)N * HIDDEN);
  float* gp2  = (float*)alloc(sizeof(float) * (size_t)N * HIDDEN);
  float* gagg = (float*)alloc(sizeof(float) * (size_t)N * HIDDEN);
  float* gxf  = (float*)alloc(sizeof(float) * (size_t)N * HIDDEN);

  auto zero = [&](float* p, size_t n) {
    zero_kernel<<<dim3((unsigned)((n + 255) / 256)), dim3(256), 0, stream>>>(p, n);
  };

  zero(out, (size_t)out_size);   // energies/forces accumulate; stress/features stay zero

  // ---- forward setup ----
  edge_geom_kernel<<<dim3((unsigned)((E + 255) / 256)), dim3(256), 0, stream>>>(
      pos, esrc, edst, E, rbfh, dist, ccv);
  embed_kernel<<<dim3((unsigned)(((size_t)N * HIDDEN + 255) / 256)), dim3(256), 0, stream>>>(
      emb, z, N, hbuf);

  const dim3 gE((unsigned)((E + 63) / 64)), gN((unsigned)((N + 63) / 64)), blk(128);
  const dim3 gEF((unsigned)(((size_t)E * HIDDEN + 255) / 256)), t256(256);

  typedef _Float16 h16;

  // ---- forward layers ----
  for (int l = 0; l < LLAYERS; ++l) {
    float* hl  = hbuf + (size_t)l * N * HIDDEN;
    float* hn  = hbuf + (size_t)(l + 1) * N * HIDDEN;
    float* xfl = xfb  + (size_t)l * N * HIDDEN;
    float* vl  = vb   + (size_t)l * N * HIDDEN;
    gemm_wmma<2,8,(F_BIAS|F_SSP),h16,h16,float><<<gE, blk, 0, stream>>>(
        rbfh, KG, w1f[l], b1 + (size_t)l*HIDDEN, nullptr, nullptr, nullptr, et1h, E);
    gemm_wmma<4,8,F_BIAS,h16,h16,float><<<gE, blk, 0, stream>>>(
        et1h, HIDDEN, w2f[l], b2 + (size_t)l*HIDDEN, nullptr, nullptr, nullptr, et2h, E);
    gemm_wmma<4,8,0,float,float,float><<<gN, blk, 0, stream>>>(
        hl, HIDDEN, l1f[l], nullptr, nullptr, nullptr, nullptr, xfl, N);
    zero(agg, (size_t)N * HIDDEN);
    scatter_msg_kernel<<<gEF, t256, 0, stream>>>(xfl, et2h, ccv, esrc, edst, E, agg);
    gemm_wmma<4,8,(F_BIAS|F_SSP),float,float,float><<<gN, blk, 0, stream>>>(
        agg, HIDDEN, l2f[l], l2b + (size_t)l*HIDDEN, nullptr, nullptr, nullptr, vl, N);
    gemm_wmma<4,8,(F_BIAS|F_RESID),float,float,float><<<gN, blk, 0, stream>>>(
        vl, HIDDEN, pf[l], pb + (size_t)l*HIDDEN, hl, nullptr, nullptr, hn, N);
  }

  // ---- head + energies ----
  float* h6 = hbuf + (size_t)LLAYERS * N * HIDDEN;
  gemm_wmma<4,4,(F_BIAS|F_SSP),float,float,float><<<gN, blk, 0, stream>>>(
      h6, HIDDEN, h1f, h1b, nullptr, nullptr, nullptr, ub, N);
  energy_kernel<<<dim3((unsigned)((N + 255) / 256)), t256, 0, stream>>>(
      ub, h2w, h2b, batch, N, out_energy);

  // ---- backward head: g_h6 = (head2_w * dssp(u)) @ head1^T ----
  ghead_kernel<<<dim3((unsigned)(((size_t)N * 64 + 255) / 256)), t256, 0, stream>>>(
      ub, h2w, N, ghp);
  gemm_wmma<2,8,0,float,float,float><<<gN, blk, 0, stream>>>(
      ghp, 64, h1t, nullptr, nullptr, nullptr, nullptr, gh, N);

  // ---- backward layers ----
  for (int l = LLAYERS - 1; l >= 0; --l) {
    float* xfl = xfb + (size_t)l * N * HIDDEN;
    float* vl  = vb  + (size_t)l * N * HIDDEN;
    // g_p2 = (g_h @ post^T) * dssp(v)
    gemm_wmma<4,8,F_AUXDSSP,float,float,float><<<gN, blk, 0, stream>>>(
        gh, HIDDEN, pt[l], nullptr, nullptr, vl, nullptr, gp2, N);
    // g_agg = g_p2 @ lin2^T
    gemm_wmma<4,8,0,float,float,float><<<gN, blk, 0, stream>>>(
        gp2, HIDDEN, l2t[l], nullptr, nullptr, nullptr, nullptr, gagg, N);
    // recompute t1 / t2 from cached rbf
    gemm_wmma<2,8,(F_BIAS|F_SSP),h16,h16,float><<<gE, blk, 0, stream>>>(
        rbfh, KG, w1f[l], b1 + (size_t)l*HIDDEN, nullptr, nullptr, nullptr, et1h, E);
    gemm_wmma<4,8,F_BIAS,h16,h16,float><<<gE, blk, 0, stream>>>(
        et1h, HIDDEN, w2f[l], b2 + (size_t)l*HIDDEN, nullptr, nullptr, nullptr, et2h, E);
    // edge backward: g_W, g_xf
    zero(gxf, (size_t)N * HIDDEN);
    scatter_bwd_kernel<<<gEF, t256, 0, stream>>>(gagg, et2h, ccv, xfl, esrc, edst, E, gxf, gWh);
    // g_cc = sum_f g_W * t2
    gcc_kernel<<<dim3((unsigned)((E + 3) / 4)), blk, 0, stream>>>(gWh, et2h, E, gcc);
    // g_t1 = ((g_W * cc) @ w2^T) * dssp(t1)   (overwrite et2)
    gemm_wmma<4,8,(F_AUXDSSP|F_ROWSCALE),h16,h16,h16><<<gE, blk, 0, stream>>>(
        gWh, HIDDEN, w2t[l], nullptr, nullptr, et1h, ccv, et2h, E);
    // g_rbf = g_t1 @ w1^T
    gemm_wmma<4,4,0,h16,h16,float><<<gE, blk, 0, stream>>>(
        et2h, HIDDEN, w1t[l], nullptr, nullptr, nullptr, nullptr, grbfh, E);
    // g_dist -> forces
    edge_force_kernel<<<dim3((unsigned)((E + 255) / 256)), t256, 0, stream>>>(
        grbfh, rbfh, dist, gcc, pos, esrc, edst, E, out_force);
    // g_h(l) = g_h(l+1) + g_xf @ lin1^T   (in-place residual)
    gemm_wmma<4,8,F_RESID,float,float,float><<<gN, blk, 0, stream>>>(
        gxf, HIDDEN, l1t[l], nullptr, gh, nullptr, nullptr, gh, N);
  }
}
